// Net_85521388798587
// MI455X (gfx1250) — matrix operand order repaired and checked by reading, not compiled, not run on hardware
//
#include <hip/hip_runtime.h>

#define NN 51200
#define EE 512000
#define E2 (EE + NN)
#define GG 512
#define SS 100
#define HH 4
#define DD 64
#define HDX 256
#define CC 10
#define NEGF (-1e30f)

typedef __attribute__((ext_vector_type(16))) __bf16 v16bf;
typedef __attribute__((ext_vector_type(8)))  __bf16 v8bf;
typedef __attribute__((ext_vector_type(8)))  float  v8f;

// ---------------- helpers ----------------

__device__ __forceinline__ __bf16 f2bf(float f) {
  union { float f; unsigned u; } in; in.f = f;
  unsigned r = in.u + 0x7FFFu + ((in.u >> 16) & 1u);   // round-to-nearest-even
  union { unsigned short s; __bf16 b; } out; out.s = (unsigned short)(r >> 16);
  return out.b;
}

// monotone encoding of float for unsigned atomicMax
__device__ __forceinline__ unsigned encf(float f) {
  unsigned u = __float_as_uint(f);
  return (u & 0x80000000u) ? ~u : (u | 0x80000000u);
}
__device__ __forceinline__ float decf(unsigned e) {
  unsigned u = (e & 0x80000000u) ? (e & 0x7FFFFFFFu) : ~e;
  return __uint_as_float(u);
}

// ---------------- conversion kernels (run once per operand) ----------------

__global__ void k_cvtbf(const float* __restrict__ s, __bf16* __restrict__ d, int n) {
  int t = blockIdx.x * blockDim.x + threadIdx.x;
  if (t < n) d[t] = f2bf(s[t]);
}

// W[K x M] f32 -> Wt[M x K] bf16 (transposed so B fragments are contiguous)
__global__ void k_cvtwt(const float* __restrict__ W, __bf16* __restrict__ Wt, int K, int M) {
  int t = blockIdx.x * blockDim.x + threadIdx.x;
  if (t >= M * K) return;
  int m = t / K, k = t - m * K;
  Wt[t] = f2bf(W[(size_t)k * M + m]);
}

// ---------------- WMMA GEMM ----------------
// Y[rows x M] = Xb[rows x K](bf16) @ Wt[M x K](bf16, transposed) (+bias)(+relu)
// one wave per 16x32 output tile (2 column tiles), K-steps of 32, f32 accumulate.
// A fragment = two contiguous b128 loads per lane; B fragment = one contiguous 32B load.
__global__ __launch_bounds__(256) void k_gemm(const __bf16* __restrict__ Xb,
                                              const __bf16* __restrict__ Wt,
                                              const float* __restrict__ bias,
                                              float* __restrict__ Y,
                                              int rows, int K, int M, int relu)
{
  int wave = threadIdx.x >> 5;
  int lane = threadIdx.x & 31;
  int row0 = (blockIdx.y * 8 + wave) * 16;
  int col0 = blockIdx.x * 32;
  if (row0 >= rows) return;           // wave-uniform; EXEC stays all-ones for WMMA
  int half = lane >> 4;
  int l16  = lane & 15;

  v8f acc0 = {}, acc1 = {};
  const __bf16* xr = Xb + (size_t)(row0 + l16) * K + half * 16;
  const __bf16* w0 = Wt + (size_t)(col0 + l16) * K + half * 16;
  const __bf16* w1 = w0 + (size_t)16 * K;

  for (int kb = 0; kb < K; kb += 32) {
    union { v16bf v; v8bf p[2]; } a, b0, b1;
    a.p[0] = *(const v8bf*)(xr + kb);        // K = kb+half*16 .. +8
    a.p[1] = *(const v8bf*)(xr + kb + 8);    // K = kb+half*16+8 .. +8
    b0.v   = *(const v16bf*)(w0 + kb);       // K = kb+half*16 .. +16, col cc0
    b1.v   = *(const v16bf*)(w1 + kb);       // same K range, col cc1
    acc0 = __builtin_amdgcn_wmma_f32_16x16x32_bf16(false, a.v, false, b0.v, (short)0, acc0,
                                                   false, false);
    acc1 = __builtin_amdgcn_wmma_f32_16x16x32_bf16(false, a.v, false, b1.v, (short)0, acc1,
                                                   false, false);
  }

  // C/D layout: VGPR r holds M-row = r + 8*half, N-col = l16
  int cc0 = col0 + l16, cc1 = cc0 + 16;
  float bv0 = bias ? bias[cc0] : 0.f;
  float bv1 = bias ? bias[cc1] : 0.f;
#pragma unroll
  for (int r = 0; r < 8; ++r) {
    size_t rr = (size_t)(row0 + r + 8 * half) * M;
    float v0 = acc0[r] + bv0;
    float v1 = acc1[r] + bv1;
    if (relu) { v0 = fmaxf(v0, 0.f); v1 = fmaxf(v1, 0.f); }
    Y[rr + cc0] = v0;
    Y[rr + cc1] = v1;
  }
}

// ---------------- utility kernels ----------------

__global__ void k_zero(float* p, int n) {
  int t = blockIdx.x * blockDim.x + threadIdx.x;
  if (t < n) p[t] = 0.f;
}
__global__ void k_filli(int* p, int v, int n) {
  int t = blockIdx.x * blockDim.x + threadIdx.x;
  if (t < n) p[t] = v;
}

// a_src[n,h] = sum_d h[n,h,d]*att[0,h,d]; a_dst with att[1]
__global__ void k_att_proj(const float* __restrict__ h, const float* __restrict__ att,
                           float* __restrict__ asrc, float* __restrict__ adst)
{
  int t = blockIdx.x * blockDim.x + threadIdx.x;
  if (t >= NN * HH) return;
  int node = t >> 2, hh = t & 3;
  const float* hp = h + node * HDX + hh * DD;
  const float* a0 = att + hh * DD;
  const float* a1 = att + HDX + hh * DD;
  float s0 = 0.f, s1 = 0.f;
  for (int d = 0; d < DD; ++d) { float v = hp[d]; s0 += v * a0[d]; s1 += v * a1[d]; }
  asrc[t] = s0; adst[t] = s1;
}

// GAT logits over E real edges + N self loops; masked -> NEG; segment max via atomicMax(enc)
__global__ void k_gat_logit(const float* __restrict__ asrc, const float* __restrict__ adst,
                            const int* __restrict__ src, const int* __restrict__ dst,
                            const int* __restrict__ emask, const int* __restrict__ nmask,
                            float* __restrict__ exb, unsigned* __restrict__ menc)
{
  int t = blockIdx.x * blockDim.x + threadIdx.x;
  if (t >= E2 * HH) return;
  int e = t >> 2, hh = t & 3;
  int s, d, msk;
  if (e < EE) { s = src[e]; d = dst[e]; msk = emask[e]; }
  else        { s = d = e - EE; msk = nmask[s]; }
  float lg;
  if (msk) { float v = asrc[s * HH + hh] + adst[d * HH + hh]; lg = v > 0.f ? v : 0.2f * v; }
  else     lg = NEGF;
  exb[t] = lg;
  atomicMax(&menc[d * HH + hh], encf(lg));
}

// GT logits over E edges: (q[dst].k[src])/sqrt(D); masked -> NEG
__global__ void k_gt_logit(const float* __restrict__ qkv,
                           const int* __restrict__ src, const int* __restrict__ dst,
                           const int* __restrict__ emask,
                           float* __restrict__ exb, unsigned* __restrict__ menc)
{
  int t = blockIdx.x * blockDim.x + threadIdx.x;
  if (t >= EE * HH) return;
  int e = t >> 2, hh = t & 3;
  int s = src[e], d = dst[e];
  float lg;
  if (emask[e]) {
    const float* q = qkv + (size_t)d * 768 + hh * DD;          // qkv[:,0]
    const float* k = qkv + (size_t)s * 768 + 256 + hh * DD;    // qkv[:,1]
    float a = 0.f;
    for (int i = 0; i < DD; ++i) a += q[i] * k[i];
    lg = a * 0.125f;
  } else lg = NEGF;
  exb[t] = lg;
  atomicMax(&menc[d * HH + hh], encf(lg));
}

// exp(lg - m) masked, accumulate segment sums
__global__ void k_exp_sum(float* __restrict__ exb, const unsigned* __restrict__ menc,
                          float* __restrict__ ssum, const int* __restrict__ dst,
                          const int* __restrict__ emask, const int* __restrict__ nmask,
                          int nE)
{
  int t = blockIdx.x * blockDim.x + threadIdx.x;
  if (t >= nE * HH) return;
  int e = t >> 2, hh = t & 3;
  int d, msk;
  if (e < EE) { d = dst[e]; msk = emask[e]; }
  else        { d = e - EE; msk = nmask[d]; }
  float ex = 0.f;
  if (msk) ex = expf(exb[t] - decf(menc[d * HH + hh]));
  exb[t] = ex;
  if (ex != 0.f) atomicAdd(&ssum[d * HH + hh], ex);
}

// out[dst,h,:] += V[src,h,:] * alpha   (V row stride / column offset parameterized)
__global__ void k_agg(const float* __restrict__ exb, const float* __restrict__ ssum,
                      const float* __restrict__ V, int vstride, int voff,
                      const int* __restrict__ src, const int* __restrict__ dst,
                      float* __restrict__ agg, int nE)
{
  int t = blockIdx.x * blockDim.x + threadIdx.x;
  if (t >= nE * HH) return;
  float ex = exb[t];
  if (ex == 0.f) return;
  int e = t >> 2, hh = t & 3;
  int s, d;
  if (e < EE) { s = src[e]; d = dst[e]; }
  else        { s = d = e - EE; }
  float alpha = ex / fmaxf(ssum[d * HH + hh], 1e-16f);
  const float* vp = V + (size_t)s * vstride + voff + hh * DD;
  float* op = agg + (size_t)d * HDX + hh * DD;
  for (int i = 0; i < DD; ++i) atomicAdd(&op[i], vp[i] * alpha);
}

// x_new = nmask ? relu(agg + skip) : 0
__global__ void k_combine(const float* __restrict__ agg, const float* __restrict__ xres,
                          const int* __restrict__ nmask, float* __restrict__ out)
{
  int t = blockIdx.x * blockDim.x + threadIdx.x;
  if (t >= NN * HDX) return;
  int n = t >> 8;
  out[t] = nmask[n] ? fmaxf(agg[t] + xres[t], 0.f) : 0.f;
}

// 1/||w|| for 256-elem pool weight
__global__ void k_norm(const float* __restrict__ w, float* __restrict__ invn) {
  __shared__ float red[256];
  float v = w[threadIdx.x];
  red[threadIdx.x] = v * v;
  __syncthreads();
  for (int s = 128; s > 0; s >>= 1) {
    if (threadIdx.x < s) red[threadIdx.x] += red[threadIdx.x + s];
    __syncthreads();
  }
  if (threadIdx.x == 0) invn[0] = rsqrtf(red[0]);
}

__global__ void k_score(const float* __restrict__ x, const float* __restrict__ w,
                        const float* __restrict__ invn, float* __restrict__ score)
{
  int t = blockIdx.x * blockDim.x + threadIdx.x;
  if (t >= NN) return;
  const float* xr = x + (size_t)t * HDX;
  float a = 0.f;
  for (int i = 0; i < HDX; ++i) a += xr[i] * w[i];
  score[t] = tanhf(a * invn[0]);
}

// top-k keep per contiguous 100-node graph; stable-tie rank matches jnp.argsort
__global__ void k_pool(const float* __restrict__ score, int* __restrict__ nmask) {
  int g = blockIdx.x, tid = threadIdx.x;
  __shared__ float ck[SS];
  __shared__ int nm_s[SS];
  __shared__ int alive;
  if (tid == 0) alive = 0;
  __syncthreads();
  int n = g * SS + tid;
  if (tid < SS) {
    int nm = nmask[n];
    nm_s[tid] = nm;
    float sc = score[n];
    ck[tid] = nm ? -fminf(fmaxf(sc, -1.5f), 1.5f) : 1.5f;  // -clip(keyv); dead nodes sort last
    if (nm) atomicAdd(&alive, 1);
  }
  __syncthreads();
  if (tid < SS) {
    int k = (int)ceilf(0.8f * (float)alive);
    float ci = ck[tid];
    int rank = 0;
    for (int j = 0; j < SS; ++j) {
      float cj = ck[j];
      if (cj < ci || (cj == ci && j < tid)) rank++;
    }
    nmask[n] = (rank < k) && nm_s[tid];
  }
}

__global__ void k_scale(float* __restrict__ x, const float* __restrict__ score,
                        const int* __restrict__ nmask)
{
  int t = blockIdx.x * blockDim.x + threadIdx.x;
  if (t >= NN * HDX) return;
  int n = t >> 8;
  x[t] = nmask[n] ? x[t] * score[n] : 0.f;
}

__global__ void k_emask(int* __restrict__ emask, const int* __restrict__ src,
                        const int* __restrict__ dst, const int* __restrict__ nmask)
{
  int t = blockIdx.x * blockDim.x + threadIdx.x;
  if (t >= EE) return;
  emask[t] = emask[t] && nmask[src[t]] && nmask[dst[t]];
}

// per-graph [max | mean] of masked x, accumulated into hcat[g, colOff .. colOff+512)
__global__ void k_readout(const float* __restrict__ x, const int* __restrict__ nmask,
                          float* __restrict__ hcat, int colOff)
{
  int g = blockIdx.x, c = threadIdx.x;  // 256 threads
  float mx = NEGF, sm = 0.f;
  int cnt = 0;
  for (int j = 0; j < SS; ++j) {
    int n = g * SS + j;
    if (nmask[n]) { float v = x[(size_t)n * HDX + c]; mx = fmaxf(mx, v); sm += v; cnt++; }
  }
  float* row = hcat + (size_t)g * 1024 + colOff;
  row[c]       += mx;
  row[256 + c] += sm / fmaxf((float)cnt, 1.f);
}

// lin3 + log_softmax, one thread per graph (C=10 too small for tiles)
__global__ void k_head(const float* __restrict__ t2, const float* __restrict__ W,
                       const float* __restrict__ b, float* __restrict__ out)
{
  int g = blockIdx.x * blockDim.x + threadIdx.x;
  if (g >= GG) return;
  const float* r = t2 + (size_t)g * 64;
  float z[CC];
  for (int c = 0; c < CC; ++c) {
    float a = b[c];
    for (int k = 0; k < 64; ++k) a += r[k] * W[k * CC + c];
    z[c] = a;
  }
  float m = z[0];
  for (int c = 1; c < CC; ++c) m = fmaxf(m, z[c]);
  float s = 0.f;
  for (int c = 0; c < CC; ++c) s += expf(z[c] - m);
  float l = logf(s);
  for (int c = 0; c < CC; ++c) out[(size_t)g * CC + c] = z[c] - m - l;
}

// ---------------- orchestration ----------------

extern "C" void kernel_launch(void* const* d_in, const int* in_sizes, int n_in,
                              void* d_out, int out_size, void* d_ws, size_t ws_size,
                              hipStream_t stream)
{
  (void)in_sizes; (void)n_in; (void)out_size; (void)ws_size;
  const float* x0  = (const float*)d_in[0];
  const int*   src = (const int*)d_in[1];
  const int*   dst = src + EE;

  float* F = (float*)d_ws;
  size_t o = 0;
  float* xcur = F + o; o += (size_t)NN * HDX;      // current branch features (f32)
  float* big  = F + o; o += (size_t)NN * 768;      // h (GAT) / qkv (GT)
  float* xres = F + o; o += (size_t)NN * HDX;      // skip-path result
  float* agg  = F + o; o += (size_t)NN * HDX;      // attention aggregation
  float* asrc = F + o; o += (size_t)NN * HH;
  float* adst = F + o; o += (size_t)NN * HH;
  float* exb  = F + o; o += (size_t)E2 * HH;       // per-edge logits/exp
  unsigned* menc = (unsigned*)(F + o); o += (size_t)NN * HH;  // encoded segment max
  float* ssum = F + o; o += (size_t)NN * HH;       // segment sums
  float* scor = F + o; o += (size_t)NN;
  float* invn = F + o; o += 64;
  int* nmask  = (int*)(F + o); o += (size_t)NN;
  int* emask  = (int*)(F + o); o += (size_t)EE;
  float* hcat = F + o; o += (size_t)GG * 1024;     // concat(h1,h2) readout accumulator
  float* t1   = F + o; o += (size_t)GG * 128;
  float* t2   = F + o; o += (size_t)GG * 64;
  __bf16* xbf = (__bf16*)(F + o); o += (size_t)NN * HDX / 2;   // bf16 activations
  __bf16* wtb = (__bf16*)(F + o); o += (size_t)768 * 256 / 2;  // bf16 transposed weight

  auto gemm = [&](const __bf16* X, const __bf16* Wt, const float* B, float* Y,
                  int rows, int K, int M, int relu) {
    dim3 grid(M / 32, rows / 128);
    k_gemm<<<grid, 256, 0, stream>>>(X, Wt, B, Y, rows, K, M, relu);
  };
  auto cvt = [&](const float* s, __bf16* d, int n) {
    k_cvtbf<<<(n + 255) / 256, 256, 0, stream>>>(s, d, n);
  };
  auto cvtw = [&](const float* W, int K, int M) {
    k_cvtwt<<<(K * M + 255) / 256, 256, 0, stream>>>(W, wtb, K, M);
  };
  auto zero = [&](float* p, size_t n) {
    k_zero<<<(unsigned)((n + 255) / 256), 256, 0, stream>>>(p, (int)n);
  };

  zero(hcat, (size_t)GG * 1024);

  for (int br = 0; br < 2; ++br) {
    k_filli<<<(NN + 255) / 256, 256, 0, stream>>>(nmask, 1, NN);
    k_filli<<<(EE + 255) / 256, 256, 0, stream>>>(emask, 1, EE);
    for (int L = 0; L < 3; ++L) {
      const float* Xin = (L == 0) ? x0 : xcur;
      int K = (L == 0) ? 64 : 256;
      int base = 3 + L * 8;
      cvt(Xin, xbf, NN * K);                       // activations -> bf16 (once per layer)
      zero((float*)menc, (size_t)NN * HH);
      zero(ssum, (size_t)NN * HH);
      zero(agg, (size_t)NN * HDX);
      if (br == 0) {  // GAT
        cvtw((const float*)d_in[base + 0], K, HDX);
        gemm(xbf, wtb, nullptr, big, NN, K, HDX, 0);
        cvtw((const float*)d_in[base + 3], K, HDX);
        gemm(xbf, wtb, (const float*)d_in[base + 2], xres, NN, K, HDX, 0);
        k_att_proj<<<(NN * HH + 255) / 256, 256, 0, stream>>>(big, (const float*)d_in[base + 1], asrc, adst);
        k_gat_logit<<<(E2 * HH + 255) / 256, 256, 0, stream>>>(asrc, adst, src, dst, emask, nmask, exb, menc);
        k_exp_sum<<<(E2 * HH + 255) / 256, 256, 0, stream>>>(exb, menc, ssum, dst, emask, nmask, E2);
        k_agg<<<(E2 * HH + 255) / 256, 256, 0, stream>>>(exb, ssum, big, HDX, 0, src, dst, agg, E2);
      } else {        // Graph Transformer
        cvtw((const float*)d_in[base + 4], K, 768);
        gemm(xbf, wtb, (const float*)d_in[base + 5], big, NN, K, 768, 0);
        cvtw((const float*)d_in[base + 6], K, HDX);
        gemm(xbf, wtb, (const float*)d_in[base + 7], xres, NN, K, HDX, 0);
        k_gt_logit<<<(EE * HH + 255) / 256, 256, 0, stream>>>(big, src, dst, emask, exb, menc);
        k_exp_sum<<<(EE * HH + 255) / 256, 256, 0, stream>>>(exb, menc, ssum, dst, emask, nmask, EE);
        k_agg<<<(EE * HH + 255) / 256, 256, 0, stream>>>(exb, ssum, big, 768, 512, src, dst, agg, EE);
      }
      k_combine<<<(NN * HDX + 255) / 256, 256, 0, stream>>>(agg, xres, nmask, xcur);
      // top-k pooling + readout
      const float* pw = (const float*)d_in[br == 0 ? 27 : 28] + L * HDX;
      k_norm<<<1, 256, 0, stream>>>(pw, invn);
      k_score<<<(NN + 255) / 256, 256, 0, stream>>>(xcur, pw, invn, scor);
      k_pool<<<GG, 128, 0, stream>>>(scor, nmask);
      k_scale<<<(NN * HDX + 255) / 256, 256, 0, stream>>>(xcur, scor, nmask);
      k_emask<<<(EE + 255) / 256, 256, 0, stream>>>(emask, src, dst, nmask);
      k_readout<<<GG, 256, 0, stream>>>(xcur, nmask, hcat, br * 512);
    }
  }

  // MLP head (WMMA): 1024->128 relu, 128->64 relu, then 64->10 + log_softmax
  cvt(hcat, xbf, GG * 1024);
  cvtw((const float*)d_in[29], 1024, 128);
  gemm(xbf, wtb, (const float*)d_in[30], t1, GG, 1024, 128, 1);
  cvt(t1, xbf, GG * 128);
  cvtw((const float*)d_in[31], 128, 64);
  gemm(xbf, wtb, (const float*)d_in[32], t2, GG, 128, 64, 1);
  k_head<<<(GG + 255) / 256, 256, 0, stream>>>(t2, (const float*)d_in[33], (const float*)d_in[34],
                                               (float*)d_out);
}
